// MPNN_30064771072051
// MI455X (gfx1250) — matrix operand order repaired and checked by reading, not compiled, not run on hardware
//
#include <hip/hip_runtime.h>

typedef __attribute__((ext_vector_type(16))) __bf16 v16bf;
typedef __attribute__((ext_vector_type(8)))  float  v8f;
typedef __attribute__((ext_vector_type(4)))  float  f32x4;
typedef __attribute__((ext_vector_type(4)))  unsigned int u32x4;
typedef __attribute__((ext_vector_type(4)))  int  i32x4;
typedef __attribute__((ext_vector_type(8)))  int  i32x8;

#define BNSC 0.99999500003750f   // 1/sqrt(1+1e-5)

#if defined(__has_builtin)
#if __has_builtin(__builtin_amdgcn_tensor_load_to_lds) && __has_builtin(__builtin_amdgcn_s_wait_tensorcnt)
#define HAVE_TDM 1
#endif
#endif
#ifndef HAVE_TDM
#define HAVE_TDM 0
#endif

__device__ __forceinline__ unsigned short f2bf(float f) {
  unsigned int u = __float_as_uint(f);
  u += 0x7fffu + ((u >> 16) & 1u);          // round-to-nearest-even
  return (unsigned short)(u >> 16);
}
__device__ __forceinline__ unsigned pack2(float a, float b) {
  return (unsigned)f2bf(a) | ((unsigned)f2bf(b) << 16);
}

union Frag16 {
  unsigned int d[8];
  v16bf v;
};

#if HAVE_TDM
// ---- Tensor Data Mover: 2D tile (global -> LDS), bf16 elements ----
// dim0/dim1 = remaining tensor extents from tile start (OOB -> zero-fill),
// tile0 x tile1 elements, stride0 = elements between rows.
__device__ __forceinline__ void tdm_load_2d(unsigned lds_off, const void* gptr,
                                            long long dim0, long long dim1,
                                            unsigned tile0, unsigned tile1,
                                            long long stride0) {
  unsigned long long ga = (unsigned long long)(size_t)gptr;
  u32x4 g0;
  g0.x = 1u;                                    // count=1, user descriptor
  g0.y = lds_off;                               // LDS byte address
  g0.z = (unsigned)ga;                          // global_addr[31:0]
  g0.w = (unsigned)(ga >> 32) | (2u << 30);     // global_addr[56:32] | type=2
  unsigned td0 = (unsigned)dim0, td1 = (unsigned)dim1;
  unsigned long long s0 = (unsigned long long)stride0;
  i32x8 g1;
  g1[0] = (int)(1u << 16);                          // data_size=1 -> 2 bytes
  g1[1] = (int)((td0 & 0xFFFFu) << 16);             // tensor_dim0[15:0]
  g1[2] = (int)((td0 >> 16) | ((td1 & 0xFFFFu) << 16));
  g1[3] = (int)((td1 >> 16) | (tile0 << 16));       // tile_dim0
  g1[4] = (int)(tile1 & 0xFFFFu);                   // tile_dim1 (tile_dim2=0)
  g1[5] = (int)(unsigned)s0;                        // tensor_dim0_stride[31:0]
  g1[6] = (int)(unsigned)((s0 >> 32) & 0xFFFFu);    // stride[47:32]
  g1[7] = 0;
  i32x4 z4 = {0, 0, 0, 0};
#if __clang_major__ >= 23
  i32x8 z8 = {0, 0, 0, 0, 0, 0, 0, 0};
  __builtin_amdgcn_tensor_load_to_lds(g0, g1, z4, z4, z8, 0);
#else
  __builtin_amdgcn_tensor_load_to_lds(g0, g1, z4, z4, 0);
#endif
}
#endif

// ---------------- A-tile loaders: 8 contiguous bf16 per call ----------------
struct ALoadHyper {               // A[h][k] = E[hyperedge[h,1+k/200]][k%200] * pos[k]
  const float* E; const int* he; const float* pos;
  static constexpr bool kTDM = false;
  __device__ u32x4 load8(int h, int k) const {
    int s = k / 200, d = k - s * 200;             // 8-chunks never straddle (200%8==0)
    int e = he[h * 7 + 1 + s];
    const f32x4* Ep = (const f32x4*)(E + (size_t)e * 200 + d);
    const f32x4* Pp = (const f32x4*)(pos + k);
    f32x4 a = Ep[0] * Pp[0];
    f32x4 b = Ep[1] * Pp[1];
    u32x4 r;
    r.x = pack2(a.x, a.y); r.y = pack2(a.z, a.w);
    r.z = pack2(b.x, b.y); r.w = pack2(b.z, b.w);
    return r;
  }
  __device__ const unsigned short* tdm_base() const { return nullptr; }
  __device__ int tdm_lda() const { return 0; }
};
struct ALoadBf16 {                // plain bf16 row-major matrix (TDM-able)
  const unsigned short* A; int lda;
  static constexpr bool kTDM = true;
  __device__ u32x4 load8(int r, int k) const {
    return *(const u32x4*)(A + (size_t)r * lda + k);
  }
  __device__ const unsigned short* tdm_base() const { return A; }
  __device__ int tdm_lda() const { return lda; }
};
struct ALoadMsg {                 // [hemb[edge_type] | E[dst]], K=400
  const unsigned short* hemb; const unsigned short* Eb;
  const int* et; const int* dst;
  static constexpr bool kTDM = false;
  __device__ u32x4 load8(int e, int k) const {
    const unsigned short* p = (k < 200) ? hemb + (size_t)et[e] * 200 + k
                                        : Eb + (size_t)dst[e] * 200 + (k - 200);
    return *(const u32x4*)p;
  }
  __device__ const unsigned short* tdm_base() const { return nullptr; }
  __device__ int tdm_lda() const { return 0; }
};

// ---------------- epilogues ----------------
struct EpHemb {                   // 0.5*acc + 0.5*R[hyperedge[r,0]] -> bf16
  unsigned short* out; const float* R; const int* he;
  __device__ void store(int r, int c, float v) const {
    float rv = R[(size_t)he[r * 7] * 200 + c];
    out[(size_t)r * 200 + c] = f2bf(0.5f * v + 0.5f * rv);
  }
};
struct EpStoreBf16 {
  unsigned short* out;
  __device__ void store(int r, int c, float v) const {
    out[(size_t)r * 200 + c] = f2bf(v);
  }
};
struct EpScatterAdd {             // segment_sum via f32 atomics
  float* agg; const int* src;
  __device__ void store(int r, int c, float v) const {
    atomicAdd(&agg[(size_t)src[r] * 200 + c], v);
  }
};
struct EpStoreF32 {
  float* out;
  __device__ void store(int r, int c, float v) const {
    out[(size_t)r * 200 + c] = v;
  }
};
struct EpFc {                     // +fc_b, bn2, relu, row-sum via atomics
  const float* fcb; const float* g2; const float* b2; float* out;
  __device__ void store(int r, int c, float v) const {
    v += fcb[c];
    v = v * (g2[c] * BNSC) + b2[c];
    v = fmaxf(v, 0.f);
    atomicAdd(&out[r], v);
  }
};

// ------- generic bf16 WMMA GEMM: C = A(MxK) * Bt^T, Bt is N x K (transposed) -------
// 256 threads = 8 waves; block tile 128(M) x 32(N); each wave owns two 16x16 C
// tiles sharing one B fragment (2 WMMA / k-step). K staged 32/step, LDS
// double-buffered; TDM tiles issued one step ahead (s_wait_tensorcnt pipelines),
// gather tiles pipelined through registers.
template <class AL, class EP>
__global__ void __launch_bounds__(256)
gemm_wmma(AL al, EP ep, const unsigned short* __restrict__ Bt,
          int M, int K, int N) {
  __shared__ unsigned short As[2 * 128 * 32];   // [buf][m][k]
  __shared__ unsigned short Bs[2 * 32 * 32];    // [buf][n][k] (B transposed)
  const int tid  = threadIdx.x;
  const int lane = tid & 31;
  const int wave = tid >> 5;
  const int mw   = wave & 3;               // 4 m-pairs
  const int nw   = wave >> 2;              // 2 n-tiles
  const int row0 = blockIdx.x * 128;
  const int n0   = blockIdx.y * 32;
  const int steps = (K + 31) >> 5;
#if HAVE_TDM
  const unsigned As_off = (unsigned)(size_t)&As[0];
  const unsigned Bs_off = (unsigned)(size_t)&Bs[0];
#endif

  v8f acc0 = {}, acc1 = {};

  // ---- fragment compute for LDS buffer (it&1): 2 WMMAs ----
  auto computeTile = [&](int it) {
    const unsigned short* Ab = &As[(it & 1) * (128 * 32)];
    const unsigned short* Bb = &Bs[(it & 1) * (32 * 32)];
    Frag16 bfr;
    {
      // B frag, contiguous k order (matches A): lane&15 = N; lanes 0-15 K=0..15, 16-31 K=16..31
      const int n  = nw * 16 + (lane & 15);
      const int kb = (lane >> 4) << 4;
      const unsigned int* Br = (const unsigned int*)&Bb[n * 32 + kb];
      #pragma unroll
      for (int t = 0; t < 8; ++t) bfr.d[t] = Br[t];
    }
    // A frag, contiguous k order (matches B): lane&15 = M; lanes 0-15 K=0..15, 16-31 K=16..31
    const int kbA = (lane >> 4) << 4;
    const unsigned short* Ar0 = &Ab[(mw * 32 + (lane & 15)) * 32];
    const unsigned short* Ar1 = Ar0 + 16 * 32;
    Frag16 af0, af1;
    #pragma unroll
    for (int t = 0; t < 8; ++t) {
      int kk = kbA + 2 * t;
      af0.d[t] = *(const unsigned int*)(Ar0 + kk);
      af1.d[t] = *(const unsigned int*)(Ar1 + kk);
    }
    acc0 = __builtin_amdgcn_wmma_f32_16x16x32_bf16(
        false, af0.v, false, bfr.v, (short)0, acc0, false, false);
    acc1 = __builtin_amdgcn_wmma_f32_16x16x32_bf16(
        false, af1.v, false, bfr.v, (short)0, acc1, false, false);
  };
  // ---- gather A tile (128 x 32) into regs: 2 chunks of 8 per thread ----
  auto fetchA = [&](int it, u32x4& r0, u32x4& r1) {
    const int k0 = it << 5;
    u32x4 z = {0u, 0u, 0u, 0u};
    r0 = z; r1 = z;
    int row = row0 + (tid >> 2), k = k0 + ((tid & 3) << 3);
    if (row < M && k < K) r0 = al.load8(row, k);
    int c1 = tid + 256;
    row = row0 + (c1 >> 2); k = k0 + ((c1 & 3) << 3);
    if (row < M && k < K) r1 = al.load8(row, k);
  };

#if HAVE_TDM
  if constexpr (AL::kTDM) {
    // -------- fully TDM-staged pipeline (A and B) --------
    if (wave == 0) {
      tdm_load_2d(As_off, al.tdm_base() + (size_t)row0 * al.tdm_lda(),
                  K, M - row0, 32u, 128u, al.tdm_lda());
      tdm_load_2d(Bs_off, Bt + (size_t)n0 * K, K, N - n0, 32u, 32u, K);
    }
    for (int it = 0; it < steps; ++it) {
      if (wave == 0) {
        if (it + 1 < steps) {
          const int k1 = (it + 1) << 5;
          tdm_load_2d(As_off + ((it + 1) & 1) * (128 * 32 * 2),
                      al.tdm_base() + (size_t)row0 * al.tdm_lda() + k1,
                      K - k1, M - row0, 32u, 128u, al.tdm_lda());
          tdm_load_2d(Bs_off + ((it + 1) & 1) * (32 * 32 * 2),
                      Bt + (size_t)n0 * K + k1, K - k1, N - n0, 32u, 32u, K);
          __builtin_amdgcn_s_wait_tensorcnt(2);   // tile `it` resident
        } else {
          __builtin_amdgcn_s_wait_tensorcnt(0);
        }
      }
      __syncthreads();
      computeTile(it);
      __syncthreads();
    }
  } else
#endif
  {
    // -------- gather A via register pipeline; B via TDM (or manual) --------
    u32x4 a0, a1;
    fetchA(0, a0, a1);
#if HAVE_TDM
    if (wave == 0)
      tdm_load_2d(Bs_off, Bt + (size_t)n0 * K, K, N - n0, 32u, 32u, K);
#else
    u32x4 b0 = {0u, 0u, 0u, 0u};
    if (tid < 128) {
      int n = n0 + (tid >> 2), k = (tid & 3) << 3;
      if (n < N && k < K) b0 = *(const u32x4*)(Bt + (size_t)n * K + k);
    }
#endif
    for (int it = 0; it < steps; ++it) {
      {   // commit staged A regs for tile `it`
        unsigned short* buf = &As[(it & 1) * (128 * 32)];
        int c1 = tid + 256;
        *(u32x4*)&buf[(tid >> 2) * 32 + ((tid & 3) << 3)] = a0;
        *(u32x4*)&buf[(c1 >> 2) * 32 + ((c1 & 3) << 3)] = a1;
      }
#if HAVE_TDM
      if (wave == 0) {
        if (it + 1 < steps) {
          const int k1 = (it + 1) << 5;
          tdm_load_2d(Bs_off + ((it + 1) & 1) * (32 * 32 * 2),
                      Bt + (size_t)n0 * K + k1, K - k1, N - n0, 32u, 32u, K);
          __builtin_amdgcn_s_wait_tensorcnt(1);   // B tile `it` resident
        } else {
          __builtin_amdgcn_s_wait_tensorcnt(0);
        }
      }
#else
      if (tid < 128) {
        unsigned short* bbuf = &Bs[(it & 1) * (32 * 32)];
        *(u32x4*)&bbuf[(tid >> 2) * 32 + ((tid & 3) << 3)] = b0;
      }
#endif
      __syncthreads();
      if (it + 1 < steps) {                       // loads overlap the WMMAs
        fetchA(it + 1, a0, a1);
#if !HAVE_TDM
        b0 = u32x4{0u, 0u, 0u, 0u};
        if (tid < 128) {
          int n = n0 + (tid >> 2), k = ((it + 1) << 5) + ((tid & 3) << 3);
          if (n < N && k < K) b0 = *(const u32x4*)(Bt + (size_t)n * K + k);
        }
#endif
      }
      computeTile(it);
      __syncthreads();
    }
  }

  // C/D layout: VGPR r -> M = r + 8*(lane>=16); N = lane&15
  const int colg = n0 + nw * 16 + (lane & 15);
  const int mhi  = (lane >> 4) << 3;
  #pragma unroll
  for (int r = 0; r < 8; ++r) {
    int rowg0 = row0 + mw * 32 + mhi + r;
    int rowg1 = rowg0 + 16;
    if (rowg0 < M && colg < N) ep.store(rowg0, colg, acc0[r]);
    if (rowg1 < M && colg < N) ep.store(rowg1, colg, acc1[r]);
  }
}

// ---------------- small elementwise kernels ----------------
__global__ void f32_to_bf16_k(const float* __restrict__ in,
                              unsigned short* __restrict__ out, int n) {
  int i = blockIdx.x * 256 + threadIdx.x;
  if (i < n) out[i] = f2bf(in[i]);
}
// convert + transpose: in is K x N fp32, out is N x K bf16
__global__ void f32_to_bf16_T_k(const float* __restrict__ in,
                                unsigned short* __restrict__ out, int K, int N) {
  int i = blockIdx.x * 256 + threadIdx.x;
  if (i >= K * N) return;
  int n = i / K, k = i - n * K;
  out[i] = f2bf(in[(size_t)k * N + n]);
}
__global__ void zero_f32_k(float* p, int n) {
  int i = blockIdx.x * 256 + threadIdx.x;
  if (i < n) p[i] = 0.f;
}
__global__ void extract_pos_k(const float* __restrict__ E, const int* pent,
                              float* __restrict__ pos, int n) {
  int i = blockIdx.x * 256 + threadIdx.x;
  if (i < n) pos[i] = E[(size_t)pent[0] * 200 + i];
}
__global__ void outE_k(const float* __restrict__ agg, const float* __restrict__ E,
                       const float* __restrict__ g, const float* __restrict__ b,
                       float* __restrict__ outE, int total) {
  int i = blockIdx.x * 256 + threadIdx.x;
  if (i >= total) return;
  int d = i % 200;
  float x = 0.5f * agg[i] + 0.5f * E[i];
  outE[i] = tanhf(x * (g[d] * BNSC) + b[d]);
}
__global__ void stk_k(const float* __restrict__ outE, const float* __restrict__ outR,
                      const float* __restrict__ ms, const int* __restrict__ ridx,
                      const int* e1, const int* e2, const int* e3,
                      const int* e4, const int* e5, const int* e6,
                      const int* pent, const float* g0, const float* b0,
                      float* __restrict__ stk, int B) {
  int i = blockIdx.x * 256 + threadIdx.x;
  if (i >= B * 200) return;
  int b = i / 200, d = i - b * 200;
  int en = pent[0];
  float gs = g0[0] * BNSC, bs = b0[0];
  stk[(size_t)b * 1400 + d] = outR[(size_t)ridx[b] * 200 + d] * gs + bs;
  const int* eidx[6] = {e1, e2, e3, e4, e5, e6};
  #pragma unroll
  for (int k = 0; k < 6; ++k) {
    float q = outE[(size_t)(en + k) * 200 + d];
    float v = outE[(size_t)eidx[k][b] * 200 + d] * q * ms[b * 6 + k];
    stk[(size_t)b * 1400 + (k + 1) * 200 + d] = v * gs + bs;
  }
}
// direct 7x3 VALID conv + bias + bn1 + relu -> bf16 (fc GEMM A input)
__global__ void conv_k(const float* __restrict__ stk, const float* __restrict__ cw,
                       const float* __restrict__ cb, const float* __restrict__ g1,
                       const float* __restrict__ b1, unsigned short* __restrict__ out,
                       int total) {
  int i = blockIdx.x * 256 + threadIdx.x;
  if (i >= total) return;
  int j = i % 198;
  int t = i / 198;
  int f = t % 200;
  int b = t / 200;
  const float* S = stk + (size_t)b * 1400;
  const float* W = cw + (size_t)f * 21;
  float s = 0.f;
  #pragma unroll
  for (int r = 0; r < 7; ++r)
    #pragma unroll
    for (int w = 0; w < 3; ++w)
      s += S[r * 200 + j + w] * W[r * 3 + w];
  s += cb[f];
  s = s * (g1[f] * BNSC) + b1[f];
  s = fmaxf(s, 0.f);
  out[(size_t)b * 39600 + (size_t)f * 198 + j] = f2bf(s);
}

// ---------------- host launch ----------------
static inline int cdiv(int a, int b) { return (a + b - 1) / b; }

extern "C" void kernel_launch(void* const* d_in, const int* in_sizes, int n_in,
                              void* d_out, int out_size, void* d_ws, size_t ws_size,
                              hipStream_t stream) {
  const float* E        = (const float*)d_in[0];
  const float* R        = (const float*)d_in[1];
  const float* w_alle   = (const float*)d_in[2];
  const float* w_addpos = (const float*)d_in[3];
  const float* w_aer    = (const float*)d_in[4];
  const float* w_rel    = (const float*)d_in[5];
  const float* conv_w   = (const float*)d_in[6];
  const float* conv_b   = (const float*)d_in[7];
  const float* fc_w     = (const float*)d_in[8];
  const float* fc_b     = (const float*)d_in[9];
  const float* bnmp_g   = (const float*)d_in[10];
  const float* bnmp_b   = (const float*)d_in[11];
  const float* bn0_g    = (const float*)d_in[12];
  const float* bn0_b    = (const float*)d_in[13];
  const float* bn1_g    = (const float*)d_in[14];
  const float* bn1_b    = (const float*)d_in[15];
  const float* bn2_g    = (const float*)d_in[16];
  const float* bn2_b    = (const float*)d_in[17];
  const float* ms       = (const float*)d_in[18];
  const int* hyperedge  = (const int*)d_in[19];
  const int* edge_index = (const int*)d_in[20];
  const int* edge_type  = (const int*)d_in[21];
  const int* r_idx      = (const int*)d_in[22];
  const int* e1 = (const int*)d_in[23];
  const int* e2 = (const int*)d_in[24];
  const int* e3 = (const int*)d_in[25];
  const int* e4 = (const int*)d_in[26];
  const int* e5 = (const int*)d_in[27];
  const int* e6 = (const int*)d_in[28];
  const int* pent = (const int*)d_in[29];

  const int D = 200;
  const int N     = in_sizes[0] / D;     // 50006
  const int NR    = in_sizes[1] / D;     // 2000
  const int H     = in_sizes[19] / 7;    // 100000
  const int NEdge = in_sizes[21];        // 600000
  const int B     = in_sizes[22];        // 2048
  const int FEAT  = 200 * 198;           // 39600
  const int* esrc = edge_index;
  const int* edst = edge_index + NEdge;

  char* base = (char*)d_ws;
  size_t off = 0;
  auto wsalloc = [&](size_t bytes) -> void* {
    void* p = base + off;
    off += (bytes + 255) & ~(size_t)255;
    return p;
  };
  unsigned short* Ebf  = (unsigned short*)wsalloc((size_t)N * D * 2);
  unsigned short* Rbf  = (unsigned short*)wsalloc((size_t)NR * D * 2);
  unsigned short* Wal  = (unsigned short*)wsalloc((size_t)6 * D * D * 2);   // 200 x 1200 (T)
  unsigned short* Waer = (unsigned short*)wsalloc((size_t)D * D * 2);       // 200 x 200 (T)
  unsigned short* Wap  = (unsigned short*)wsalloc((size_t)2 * D * D * 2);   // 200 x 400 (T)
  unsigned short* Wrl  = (unsigned short*)wsalloc((size_t)D * D * 2);       // 200 x 200 (T)
  unsigned short* Fcw  = (unsigned short*)wsalloc((size_t)FEAT * D * 2);    // 200 x 39600 (T)
  float*          posw = (float*)wsalloc((size_t)6 * D * 4);
  unsigned short* tmp1 = (unsigned short*)wsalloc((size_t)H * D * 2);
  unsigned short* hemb = (unsigned short*)wsalloc((size_t)H * D * 2);
  float*          agg  = (float*)wsalloc((size_t)N * D * 4);
  float*          outE = (float*)wsalloc((size_t)N * D * 4);
  float*          outR = (float*)wsalloc((size_t)NR * D * 4);
  float*          stk  = (float*)wsalloc((size_t)B * 7 * D * 4);
  unsigned short* cvo  = (unsigned short*)wsalloc((size_t)B * FEAT * 2);

  // dtype conversions: gather tables straight, weights converted + transposed
  f32_to_bf16_k<<<cdiv(N * D, 256), 256, 0, stream>>>(E, Ebf, N * D);
  f32_to_bf16_k<<<cdiv(NR * D, 256), 256, 0, stream>>>(R, Rbf, NR * D);
  f32_to_bf16_T_k<<<cdiv(6 * D * D, 256), 256, 0, stream>>>(w_alle, Wal, 6 * D, D);
  f32_to_bf16_T_k<<<cdiv(D * D, 256), 256, 0, stream>>>(w_aer, Waer, D, D);
  f32_to_bf16_T_k<<<cdiv(2 * D * D, 256), 256, 0, stream>>>(w_addpos, Wap, 2 * D, D);
  f32_to_bf16_T_k<<<cdiv(D * D, 256), 256, 0, stream>>>(w_rel, Wrl, D, D);
  f32_to_bf16_T_k<<<cdiv(FEAT * D, 256), 256, 0, stream>>>(fc_w, Fcw, FEAT, D);
  extract_pos_k<<<cdiv(6 * D, 256), 256, 0, stream>>>(E, pent, posw, 6 * D);
  zero_f32_k<<<cdiv(N * D, 256), 256, 0, stream>>>(agg, N * D);
  zero_f32_k<<<cdiv(B, 256), 256, 0, stream>>>((float*)d_out, B);

  const int NB = cdiv(D, 32);  // 7 n-blocks of 32 (guarded to 200)

  // G1: tmp1 = 0.5*(eh @ w_alle) + 0.5*R[he[:,0]]   (H x 1200 @ 1200 x 200)
  {
    ALoadHyper al{E, hyperedge, posw};
    EpHemb ep{tmp1, R, hyperedge};
    gemm_wmma<<<dim3(cdiv(H, 128), NB), 256, 0, stream>>>(al, ep, Wal, H, 6 * D, D);
  }
  // G2: hemb = tmp1 @ w_alleandr
  {
    ALoadBf16 al{tmp1, D};
    EpStoreBf16 ep{hemb};
    gemm_wmma<<<dim3(cdiv(H, 128), NB), 256, 0, stream>>>(al, ep, Waer, H, D, D);
  }
  // G3: msg GEMM fused with segment_sum scatter (NEdge x 400 @ 400 x 200)
  {
    ALoadMsg al{hemb, Ebf, edge_type, edst};
    EpScatterAdd ep{agg, esrc};
    gemm_wmma<<<dim3(cdiv(NEdge, 128), NB), 256, 0, stream>>>(al, ep, Wap, NEdge, 2 * D, D);
  }
  // outE = tanh(bn(0.5*agg + 0.5*E))
  outE_k<<<cdiv(N * D, 256), 256, 0, stream>>>(agg, E, bnmp_g, bnmp_b, outE, N * D);
  // G4: outR = R @ w_rel
  {
    ALoadBf16 al{Rbf, D};
    EpStoreF32 ep{outR};
    gemm_wmma<<<dim3(cdiv(NR, 128), NB), 256, 0, stream>>>(al, ep, Wrl, NR, D, D);
  }
  // stacked conv input (with bn0)
  stk_k<<<cdiv(B * D, 256), 256, 0, stream>>>(outE, outR, ms, r_idx,
                                              e1, e2, e3, e4, e5, e6,
                                              pent, bn0_g, bn0_b, stk, B);
  // conv + bias + bn1 + relu -> bf16
  conv_k<<<cdiv(B * 200 * 198, 256), 256, 0, stream>>>(stk, conv_w, conv_b,
                                                       bn1_g, bn1_b, cvo,
                                                       B * 200 * 198);
  // G5: fc + bn2 + relu + row-sum -> d_out  (B x 39600 @ 39600 x 200)
  {
    ALoadBf16 al{cvo, FEAT};
    EpFc ep{fc_b, bn2_g, bn2_b, (float*)d_out};
    gemm_wmma<<<dim3(cdiv(B, 128), NB), 256, 0, stream>>>(al, ep, Fcw, B, FEAT, D);
  }
  (void)n_in; (void)out_size; (void)ws_size; (void)conv_w;
}
